// VectorQuantizerMultiHead_79267916415516
// MI455X (gfx1250) — matrix operand order repaired and checked by reading, not compiled, not run on hardware
//
#include <hip/hip_runtime.h>
#include <hip/hip_bf16.h>
#include <stdint.h>

// ---------------------------------------------------------------------------
// VectorQuantizerMultiHead for MI455X (gfx1250, wave32)
//   N=8192 tokens, H=8 heads, K=4096 codes, d=128
//   GEMM path: v_wmma_f32_16x16x32_bf16, 3-term bf16 split (fp32-level accuracy)
//   Each wave owns 32 token rows (2 A-tiles) -> every B fragment feeds 6 WMMAs.
//   Codebook tiles staged with double-buffered GLOBAL_LOAD_ASYNC_TO_LDS_B128
//   (ASYNCcnt), overlapping DMA with the WMMA pipe; one barrier per tile.
// ---------------------------------------------------------------------------

typedef __attribute__((ext_vector_type(16))) __bf16 v16bf;
typedef __attribute__((ext_vector_type(8)))  __bf16 v8bf;
typedef __attribute__((ext_vector_type(8)))  float  v8f;

#define N_TOK   8192
#define NHEAD   8
#define KCODE   4096
#define DIM     128
#define QTOTAL  (N_TOK * NHEAD * DIM)   // 8,388,608 output elements
#define TILE_K  64                       // codebook rows staged in LDS per iter
#define NTILES  (KCODE / TILE_K)         // 64
#define LDS_STR 136                      // bf16 per LDS row: 128 + 8 pad
                                         // (272 B row stride = 68 dwords ≡ 4 mod 64 banks)

union U16x { v16bf v; v8bf h[2]; };

__device__ inline unsigned long long shfl_xor16_u64(unsigned long long v, int mask) {
  unsigned lo = (unsigned)(v & 0xFFFFFFFFull);
  unsigned hi = (unsigned)(v >> 32);
  lo = __shfl_xor(lo, mask, 16);   // stays within each 16-lane half
  hi = __shfl_xor(hi, mask, 16);
  return ((unsigned long long)hi << 32) | (unsigned long long)lo;
}

// monotone (for max) packing of (float value, code idx); ties -> smaller idx wins
__device__ inline unsigned long long pack_key(float v, unsigned idx) {
  unsigned u   = __float_as_uint(v);
  unsigned key = ((int)u >= 0) ? (u | 0x80000000u) : ~u;
  return ((unsigned long long)key << 32) | (unsigned long long)(0xFFFFFFFFu - idx);
}

// 16-byte async global->LDS copy (per lane). VDST carries the LDS byte address
// (low 32 bits of a generic shared pointer == LDS offset per aperture rules).
__device__ inline void async_ld16(unsigned lds_off, const void* gptr) {
  unsigned long long ga = (unsigned long long)(uintptr_t)gptr;
  asm volatile("global_load_async_to_lds_b128 %0, %1, off"
               :: "v"(lds_off), "v"(ga) : "memory");
}
__device__ inline void wait_async0() {
  asm volatile("s_wait_asynccnt 0" ::: "memory");
}

// --------------------------------------------------------------------------
// Kernel 1: split emb (fp32) -> bf16 hi/lo pair, and compute ||emb[h,k]||^2.
// One wave per codebook row (32 lanes x 4 floats).
// --------------------------------------------------------------------------
__global__ void vq_prep(const float* __restrict__ emb,
                        __bf16* __restrict__ ehi, __bf16* __restrict__ elo,
                        float* __restrict__ norms) {
  const int wave = threadIdx.x >> 5;
  const int lane = threadIdx.x & 31;
  const int row  = blockIdx.x * 8 + wave;          // 0 .. H*K-1 (grid sized exactly)
  const float* src = emb + (size_t)row * DIM;
  __bf16* dh = ehi + (size_t)row * DIM;
  __bf16* dl = elo + (size_t)row * DIM;
  float s = 0.f;
  #pragma unroll
  for (int i = 0; i < 4; ++i) {
    float v  = src[lane + 32 * i];
    __bf16 h = (__bf16)v;                  // RNE
    float  r = v - (float)h;               // exact residual
    dh[lane + 32 * i] = h;
    dl[lane + 32 * i] = (__bf16)r;
    s += v * v;
  }
  #pragma unroll
  for (int m = 16; m >= 1; m >>= 1) s += __shfl_xor(s, m, 32);
  if (lane == 0) norms[row] = s;
}

// --------------------------------------------------------------------------
// Kernel 2: fused GEMM + argmin.
//   grid = 64 row-tiles * 8 heads; block = 128 threads = 4 waves.
//   Wave w owns 32 rows [tile*128 + w*32, +32) of head h (two 16-row A tiles).
//   score t = 2*dot - ||e||^2 ; argmin(dist) == argmax(t).
//
// WMMA 16x16x32 bf16 operand layouts as used here (contiguous k in A and B):
//   A (16x32, MxK): lane L: m=L&15; elems 0..15 -> K=kb+i,
//                   kb = (L<16 ? 0 : 16)
//   B (32x16, KxN): lane L: n=L&15; elems 0..15 -> K=kb2+i, kb2 = (L<16 ? 0 : 16)
//   C/D (16x16 f32): vgpr j: lanes 0..15 -> M=j, lanes 16..31 -> M=j+8; N = L&15
// --------------------------------------------------------------------------
__global__ void __launch_bounds__(128) vq_argmin(
    const float*  __restrict__ x,
    const __bf16* __restrict__ ehi, const __bf16* __restrict__ elo,
    const float*  __restrict__ norms,
    int* __restrict__ codes_i, float* __restrict__ codes_f)
{
  // [buffer][hi/lo][padded tile] : 2*2*8704*2B = 69,632 B of LDS
  __shared__ __align__(16) __bf16 sbuf[2][2][TILE_K * LDS_STR];

  const int h    = blockIdx.x >> 6;     // consecutive blocks share a head -> L2 reuse
  const int tile = blockIdx.x & 63;
  const int wave = threadIdx.x >> 5;
  const int lane = threadIdx.x & 31;
  const int half = lane >> 4;
  const int m    = lane & 15;
  const int n_base = tile * 128 + wave * 32;

  const __bf16* gh = ehi + (size_t)h * KCODE * DIM;
  const __bf16* gl = elo + (size_t)h * KCODE * DIM;

  const int t      = threadIdx.x;
  const int ldrow  = t >> 1;    // 0..63 : codebook row within tile
  const int ldpart = t & 1;     // 0..1  : 64 bf16 (128B) each

  // ---- kick off async DMA of tile 0 before doing any local work ----
  {
    const __bf16* g0 = gh + (size_t)ldrow * DIM + ldpart * 64;
    const __bf16* g1 = gl + (size_t)ldrow * DIM + ldpart * 64;
    const unsigned d0 = (unsigned)(uintptr_t)(&sbuf[0][0][ldrow * LDS_STR + ldpart * 64]);
    const unsigned d1 = (unsigned)(uintptr_t)(&sbuf[0][1][ldrow * LDS_STR + ldpart * 64]);
    #pragma unroll
    for (int i = 0; i < 8; ++i) {
      async_ld16(d0 + i * 16, g0 + i * 8);
      async_ld16(d1 + i * 16, g1 + i * 8);
    }
  }

  // ---- A fragments: two 16-row tiles, bf16 hi/lo, resident in VGPRs ----
  v16bf A0hi[4], A0lo[4], A1hi[4], A1lo[4];
  #pragma unroll
  for (int rt = 0; rt < 2; ++rt) {
    const float* xrow = x + (size_t)(n_base + rt * 16 + m) * (NHEAD * DIM) + (size_t)h * DIM;
    #pragma unroll
    for (int c = 0; c < 4; ++c) {
      const int kb = c * 32 + half * 16;
      union { float4 f4[4]; float f[16]; } u;
      const float4* p = (const float4*)(xrow + kb);        // 32B aligned
      u.f4[0] = p[0]; u.f4[1] = p[1];
      p = (const float4*)(xrow + kb + 8);
      u.f4[2] = p[0]; u.f4[3] = p[1];
      v16bf th, tl;
      #pragma unroll
      for (int i = 0; i < 8; ++i) {
        __bf16 h0 = (__bf16)u.f[i];
        __bf16 h1 = (__bf16)u.f[8 + i];
        th[i]     = h0;  tl[i]     = (__bf16)(u.f[i]     - (float)h0);
        th[i + 8] = h1;  tl[i + 8] = (__bf16)(u.f[8 + i] - (float)h1);
      }
      if (rt == 0) { A0hi[c] = th; A0lo[c] = tl; }
      else         { A1hi[c] = th; A1lo[c] = tl; }
    }
  }

  // running per-lane argmax: value + index (strict '>' keeps earliest k on ties)
  float    bv0[8], bv1[8];
  unsigned bi0[8], bi1[8];
  #pragma unroll
  for (int j = 0; j < 8; ++j) {
    bv0[j] = bv1[j] = __int_as_float(0xFF800000); // -inf
    bi0[j] = bi1[j] = 0u;
  }

  for (int it = 0; it < NTILES; ++it) {
    const int p  = it & 1;
    const int k0 = it * TILE_K;

    wait_async0();        // own tile-`it` transfers have landed in LDS
    __syncthreads();      // ... block-wide; also: everyone left buf[p^1] (tile it-1)

    if (it + 1 < NTILES) {   // DMA next tile into the other buffer during compute
      const __bf16* g0 = gh + (size_t)(k0 + TILE_K + ldrow) * DIM + ldpart * 64;
      const __bf16* g1 = gl + (size_t)(k0 + TILE_K + ldrow) * DIM + ldpart * 64;
      const unsigned d0 = (unsigned)(uintptr_t)(&sbuf[p ^ 1][0][ldrow * LDS_STR + ldpart * 64]);
      const unsigned d1 = (unsigned)(uintptr_t)(&sbuf[p ^ 1][1][ldrow * LDS_STR + ldpart * 64]);
      #pragma unroll
      for (int i = 0; i < 8; ++i) {
        async_ld16(d0 + i * 16, g0 + i * 8);
        async_ld16(d1 + i * 16, g1 + i * 8);
      }
    }

    const __bf16* sh_hi = sbuf[p][0];
    const __bf16* sh_lo = sbuf[p][1];

    #pragma unroll
    for (int g = 0; g < 4; ++g) {                // 4 groups of 16 codes
      v8f acc0 = {};
      v8f acc1 = {};
      #pragma unroll
      for (int c = 0; c < 4; ++c) {              // depth chunks of 32
        const int off = (g * 16 + m) * LDS_STR + c * 32 + half * 16;
        U16x bh, bl;
        bh.h[0] = *(const v8bf*)(sh_hi + off);
        bh.h[1] = *(const v8bf*)(sh_hi + off + 8);
        bl.h[0] = *(const v8bf*)(sh_lo + off);
        bl.h[1] = *(const v8bf*)(sh_lo + off + 8);
        // dot ≈ hi*hi + hi*lo + lo*hi; interleave the two independent D-chains
        acc0 = __builtin_amdgcn_wmma_f32_16x16x32_bf16(false, A0hi[c], false, bh.v,
                                                       (short)0, acc0, false, false);
        acc1 = __builtin_amdgcn_wmma_f32_16x16x32_bf16(false, A1hi[c], false, bh.v,
                                                       (short)0, acc1, false, false);
        acc0 = __builtin_amdgcn_wmma_f32_16x16x32_bf16(false, A0hi[c], false, bl.v,
                                                       (short)0, acc0, false, false);
        acc1 = __builtin_amdgcn_wmma_f32_16x16x32_bf16(false, A1hi[c], false, bl.v,
                                                       (short)0, acc1, false, false);
        acc0 = __builtin_amdgcn_wmma_f32_16x16x32_bf16(false, A0lo[c], false, bh.v,
                                                       (short)0, acc0, false, false);
        acc1 = __builtin_amdgcn_wmma_f32_16x16x32_bf16(false, A1lo[c], false, bh.v,
                                                       (short)0, acc1, false, false);
      }
      const unsigned kidx = (unsigned)(k0 + g * 16 + m);     // this lane's column/code
      const float en = norms[(size_t)h * KCODE + kidx];
      #pragma unroll
      for (int j = 0; j < 8; ++j) {
        const float t0 = fmaf(2.0f, acc0[j], -en);
        if (t0 > bv0[j]) { bv0[j] = t0; bi0[j] = kidx; }
        const float t1 = fmaf(2.0f, acc1[j], -en);
        if (t1 > bv1[j]) { bv1[j] = t1; bi1[j] = kidx; }
      }
    }
  }

  // pack once, then reduce the 16 columns held per half-wave
  unsigned long long b0[8], b1[8];
  #pragma unroll
  for (int j = 0; j < 8; ++j) {
    b0[j] = pack_key(bv0[j], bi0[j]);
    b1[j] = pack_key(bv1[j], bi1[j]);
  }
  #pragma unroll
  for (int msk = 1; msk < 16; msk <<= 1) {
    #pragma unroll
    for (int j = 0; j < 8; ++j) {
      unsigned long long o0 = shfl_xor16_u64(b0[j], msk);
      b0[j] = b0[j] > o0 ? b0[j] : o0;
      unsigned long long o1 = shfl_xor16_u64(b1[j], msk);
      b1[j] = b1[j] > o1 ? b1[j] : o1;
    }
  }

  if (m == 0) {                      // lanes 0 and 16 emit rows j / j+8 of each tile
    #pragma unroll
    for (int j = 0; j < 8; ++j) {
      const int r  = j + 8 * half;
      const int n0 = n_base + r;
      const int n1 = n_base + 16 + r;
      const unsigned k0w = 0xFFFFFFFFu - (unsigned)(b0[j] & 0xFFFFFFFFull);
      const unsigned k1w = 0xFFFFFFFFu - (unsigned)(b1[j] & 0xFFFFFFFFull);
      codes_i[(size_t)n0 * NHEAD + h] = (int)k0w;        // [N,H] for gather
      codes_i[(size_t)n1 * NHEAD + h] = (int)k1w;
      codes_f[(size_t)h * N_TOK + n0] = (float)k0w;      // codes.T, exact in fp32
      codes_f[(size_t)h * N_TOK + n1] = (float)k1w;
    }
  }
}

// --------------------------------------------------------------------------
// Kernel 3: gather winning rows (original fp32 emb), write quantized_st,
// accumulate per-block SSE partials (deterministic tree reduction).
// --------------------------------------------------------------------------
__global__ void vq_gather(const float* __restrict__ x, const float* __restrict__ emb,
                          const int* __restrict__ codes_i,
                          float* __restrict__ out, float* __restrict__ partial) {
  const int tid = threadIdx.x;
  float s = 0.f;
  const size_t base = (size_t)blockIdx.x * 256 + tid;
  #pragma unroll
  for (int it = 0; it < 4; ++it) {
    const size_t e = base + (size_t)it * (size_t)(8192 * 256);   // covers QTOTAL exactly
    const int n = (int)(e >> 10);
    const int h = (int)((e >> 7) & 7);
    const int j = (int)(e & 127);
    const int k = codes_i[(size_t)n * NHEAD + h];
    const float q  = emb[((size_t)h * KCODE + (size_t)k) * DIM + j];
    const float xv = x[e];
    const float d  = q - xv;
    out[1 + e] = xv + d;          // inputs + stop_grad(quantized - inputs)
    s += d * d;
  }
  __shared__ float red[256];
  red[tid] = s;
  __syncthreads();
  for (int w = 128; w >= 1; w >>= 1) {
    if (tid < w) red[tid] += red[tid + w];
    __syncthreads();
  }
  if (tid == 0) partial[blockIdx.x] = red[0];
}

// Kernel 4: final deterministic loss reduction.
__global__ void vq_loss(const float* __restrict__ partial, float* __restrict__ out) {
  const int tid = threadIdx.x;
  float s = 0.f;
  for (int i = tid; i < 8192; i += 256) s += partial[i];
  __shared__ float red[256];
  red[tid] = s;
  __syncthreads();
  for (int w = 128; w >= 1; w >>= 1) {
    if (tid < w) red[tid] += red[tid + w];
    __syncthreads();
  }
  if (tid == 0) out[0] = 0.25f * (red[0] / (float)QTOTAL);
}

// --------------------------------------------------------------------------
extern "C" void kernel_launch(void* const* d_in, const int* in_sizes, int n_in,
                              void* d_out, int out_size, void* d_ws, size_t ws_size,
                              hipStream_t stream) {
  (void)in_sizes; (void)n_in; (void)out_size; (void)ws_size;
  const float* x   = (const float*)d_in[0];   // (N, H*d) fp32
  const float* emb = (const float*)d_in[1];   // (H, K, d) fp32

  // workspace layout (≈17.2 MB, 16B-aligned slots)
  char* w = (char*)d_ws;
  __bf16* ehi    = (__bf16*)(w);                              // 8,388,608 B
  __bf16* elo    = (__bf16*)(w + 8388608);                    // 8,388,608 B
  float*  norms  = (float*) (w + 16777216);                   //   131,072 B
  int*    codesI = (int*)   (w + 16777216 + 131072);          //   262,144 B
  float*  part   = (float*) (w + 16777216 + 131072 + 262144); //    32,768 B

  float* out     = (float*)d_out;            // [loss][quantized 8388608][codes 65536]
  float* codesF  = out + 1 + QTOTAL;

  vq_prep  <<<4096, 256, 0, stream>>>(emb, ehi, elo, norms);
  vq_argmin<<< 512, 128, 0, stream>>>(x, ehi, elo, norms, codesI, codesF);
  vq_gather<<<8192, 256, 0, stream>>>(x, emb, codesI, out, part);
  vq_loss  <<<   1, 256, 0, stream>>>(part, out);
}
